// GRUCell_41712722378993
// MI455X (gfx1250) — compile-verified, matrix operand order checked by reading, not run on hardware
//
#include <hip/hip_runtime.h>

// ---------------------------------------------------------------------------
// LayerNorm-GRU cell, B=65536, I=H=512
//  - bf16 WMMA (v_wmma_f32_16x16x32_bf16), f32 accumulate
//  - async global->LDS staging (global_load_async_to_lds_b128 + ASYNCcnt),
//    double-buffered B panel; whole A panel resident in LDS per block
// ---------------------------------------------------------------------------

typedef __attribute__((ext_vector_type(16))) __bf16 v16bf;
typedef __attribute__((ext_vector_type(8)))  float  v8f;

#define Bsz   65536
#define DIM   512
#define MT    32        // rows per block
#define KC    32        // K per chunk (one WMMA K)
#define LDSB  40        // B-panel row stride in bf16 (32+8 pad -> 80B, conflict-free)
#define LDSA  1032      // A-panel row stride in bf16 (1024+8 pad -> 2064B)
#define NTHR  512       // 16 waves

// --- kernel1 LDS layout (bytes) --------------------------------------------
#define K1_BBUF    81920u                 // 1024 * 40 * 2
#define K1_APAN   163840u                 // after 2 B buffers
#define K1_SMEM   229888u                 // + 32*1032*2 A panel
// --- kernel2 LDS layout ----------------------------------------------------
#define K2_BBUF    40960u                 // 512 * 40 * 2
#define K2_APAN    81920u
#define K2_SMEM   147968u

// ---- CDNA5 async copy + wait (ASYNCcnt) -----------------------------------
__device__ __forceinline__ void async_b128(unsigned lds_off, const void* g) {
  asm volatile("global_load_async_to_lds_b128 %0, %1, off"
               :: "v"(lds_off), "v"((unsigned long long)(uintptr_t)g)
               : "memory");
}
__device__ __forceinline__ void wait_async_le8() { asm volatile("s_wait_asynccnt 8" ::: "memory"); }
__device__ __forceinline__ void wait_async_le4() { asm volatile("s_wait_asynccnt 4" ::: "memory"); }
__device__ __forceinline__ void wait_async_0()   { asm volatile("s_wait_asynccnt 0" ::: "memory"); }

// ---- fragment gather from LDS, CDNA5 16-bit A/B-matrix layout --------------
// lane&15 = row (M or N); lane>>4 selects K-halves {0..7,16..23} vs {8..15,24..31}
__device__ __forceinline__ v16bf frag_ld(const __bf16* base, int stride, int lane) {
  const int r  = lane & 15;
  const int kh = (lane >> 4) << 3;
  const __bf16* p = base + r * stride;
  union { uint4 q[2]; v16bf v; } u;
  u.q[0] = *reinterpret_cast<const uint4*>(p + kh);
  u.q[1] = *reinterpret_cast<const uint4*>(p + 16 + kh);
  return u.v;
}

// ---- fill whole A panel: 32 rows x 1024 K (x | h), f32 -> bf16 -------------
__device__ __forceinline__ void fill_A_panel(__bf16* Apan, const float* x,
                                             const float* h, int mBase, int tid) {
#pragma unroll
  for (int i = 0; i < 16; ++i) {
    const int u   = i * NTHR + tid;        // 0..8191
    const int row = u >> 8;                // 0..31
    const int col = (u & 255) << 2;        // 0..1020
    const float* src = (col < DIM) ? (x + (size_t)(mBase + row) * DIM + col)
                                   : (h + (size_t)(mBase + row) * DIM + col - DIM);
    float4 f = *reinterpret_cast<const float4*>(src);
    union { __bf16 h4[4]; uint2 u2; } pk;
    pk.h4[0] = (__bf16)f.x; pk.h4[1] = (__bf16)f.y;
    pk.h4[2] = (__bf16)f.z; pk.h4[3] = (__bf16)f.w;
    *reinterpret_cast<uint2*>(Apan + row * LDSA + col) = pk.u2;
  }
}

// ---- issue one B-panel stage via async DMA (nunit = rows*4/512) ------------
__device__ __forceinline__ void stage_B(const __bf16* Wcat, unsigned ldsBase,
                                        int c, int tid, int nunit) {
#pragma unroll 8
  for (int p = 0; p < nunit; ++p) {
    const int u   = tid + p * NTHR;
    const int row = u >> 2;
    const int seg = (u & 3) << 3;          // bf16 elems 0,8,16,24
    async_b128(ldsBase + (unsigned)(row * (LDSB * 2) + seg * 2),
               Wcat + (size_t)row * 1024 + c * KC + seg);
  }
}

// ===========================================================================
// Kernel 0: pack [w_rz | u_rz] -> bf16 (1024x1024) and [w_h | u_h] (512x1024)
// ===========================================================================
__global__ void prepack_weights(const float* __restrict__ wrz,
                                const float* __restrict__ urz,
                                const float* __restrict__ wh,
                                const float* __restrict__ uh,
                                __bf16* __restrict__ Wrz,
                                __bf16* __restrict__ Wh) {
  const int t  = blockIdx.x * 256 + threadIdx.x;
  const int NW = 1024 * 1024;
  if (t < NW) {
    const int n = t >> 10, k = t & 1023;
    const float v = (k < DIM) ? wrz[n * DIM + k] : urz[n * DIM + (k - DIM)];
    Wrz[t] = (__bf16)v;
  } else {
    const int t2 = t - NW;
    const int n = t2 >> 10, k = t2 & 1023;
    const float v = (k < DIM) ? wh[n * DIM + k] : uh[n * DIM + (k - DIM)];
    Wh[t2] = (__bf16)v;
  }
}

// ===========================================================================
// Kernel 1: rz = sigmoid(LN(x@w_rz^T + b + h@u_rz^T)) -> bf16 ws [B,1024]
// ===========================================================================
__global__ __launch_bounds__(NTHR) void rz_gemm_ln(
    const float* __restrict__ x, const float* __restrict__ h,
    const __bf16* __restrict__ Wcat, const float* __restrict__ bias,
    const float* __restrict__ gam, const float* __restrict__ bet,
    __bf16* __restrict__ rz_out) {
  extern __shared__ char smem[];
  __bf16* Apan = reinterpret_cast<__bf16*>(smem + K1_APAN);
  const unsigned smemBase = (unsigned)(uintptr_t)(void*)smem;

  const int tid  = threadIdx.x;
  const int lane = tid & 31;
  const int wid  = tid >> 5;
  const int wm   = wid >> 3;          // 0..1  (16 rows)
  const int wn   = wid & 7;           // 0..7  (128 cols)
  const int mBase = blockIdx.x * MT;

  v8f acc[8];
  const v8f vzero = {};
#pragma unroll
  for (int i = 0; i < 8; ++i) acc[i] = vzero;

  fill_A_panel(Apan, x, h, mBase, tid);
  stage_B(Wcat, smemBase, 0, tid, 8);                     // stage for c=0

  for (int c = 0; c < 32; ++c) {                          // K = 1024
    if (c + 1 < 32) {
      stage_B(Wcat, smemBase + ((c + 1) & 1) * K1_BBUF, c + 1, tid, 8);
      wait_async_le8();                                   // stage c landed
    } else {
      wait_async_0();
    }
    __syncthreads();

    const v16bf af = frag_ld(Apan + (wm * 16) * LDSA + c * KC, LDSA, lane);
    const __bf16* bb = reinterpret_cast<const __bf16*>(smem + (c & 1) * K1_BBUF)
                     + (wn * 128) * LDSB;
    v16bf bf[8];
#pragma unroll
    for (int t = 0; t < 8; ++t) bf[t] = frag_ld(bb + t * 16 * LDSB, LDSB, lane);
#pragma unroll
    for (int t = 0; t < 8; ++t)
      acc[t] = __builtin_amdgcn_wmma_f32_16x16x32_bf16(
          false, af, false, bf[t], (short)0, acc[t], false, false);
    __syncthreads();
  }

  // ---- epilogue: acc(+bias) -> LDS f32, per-row LN + sigmoid --------------
  float* E = reinterpret_cast<float*>(smem);
  const int ldE = 1032;
  const int r0 = wm * 16 + ((lane >> 4) << 3);
#pragma unroll
  for (int t = 0; t < 8; ++t) {
    const int col = wn * 128 + t * 16 + (lane & 15);
    const float b = bias[col];
#pragma unroll
    for (int v = 0; v < 8; ++v)
      E[(r0 + v) * ldE + col] = acc[t][v] + b;
  }
  __syncthreads();

#pragma unroll
  for (int rr = 0; rr < 2; ++rr) {
    const int row = wid * 2 + rr;
    const size_t mg = (size_t)(mBase + row);
    float vals[32], s1 = 0.f, s2 = 0.f;
#pragma unroll
    for (int j = 0; j < 32; ++j) {
      const float v = E[row * ldE + lane + j * 32];
      vals[j] = v; s1 += v; s2 += v * v;
    }
#pragma unroll
    for (int o = 16; o; o >>= 1) {
      s1 += __shfl_xor(s1, o, 32);
      s2 += __shfl_xor(s2, o, 32);
    }
    const float mean = s1 * (1.f / 1024.f);
    const float rstd = rsqrtf(s2 * (1.f / 1024.f) - mean * mean + 1e-5f);
#pragma unroll
    for (int j = 0; j < 32; ++j) {
      const int idx = lane + j * 32;
      const float v = (vals[j] - mean) * rstd * gam[idx] + bet[idx];
      rz_out[mg * 1024 + idx] = (__bf16)(1.f / (1.f + __expf(-v)));
    }
  }
}

// ===========================================================================
// Kernel 2: h_t = (1-z)*h + z*tanh(LN(x@w_h^T + b + r*(h@u_h^T)))
// ===========================================================================
__global__ __launch_bounds__(NTHR) void h_gemm_out(
    const float* __restrict__ x, const float* __restrict__ hprev,
    const __bf16* __restrict__ Wcat, const float* __restrict__ bias,
    const float* __restrict__ gam, const float* __restrict__ bet,
    const __bf16* __restrict__ rz, float* __restrict__ out) {
  extern __shared__ char smem[];
  __bf16* Apan = reinterpret_cast<__bf16*>(smem + K2_APAN);
  const unsigned smemBase = (unsigned)(uintptr_t)(void*)smem;

  const int tid  = threadIdx.x;
  const int lane = tid & 31;
  const int wid  = tid >> 5;
  const int wm   = wid >> 3;
  const int wn   = wid & 7;           // 64 cols per wave (of 512)
  const int mBase = blockIdx.x * MT;

  v8f ax[4], ah[4];
  const v8f vzero = {};
#pragma unroll
  for (int i = 0; i < 4; ++i) { ax[i] = vzero; ah[i] = vzero; }

  fill_A_panel(Apan, x, hprev, mBase, tid);
  stage_B(Wcat, smemBase, 0, tid, 4);

  for (int c = 0; c < 32; ++c) {
    if (c + 1 < 32) {
      stage_B(Wcat, smemBase + ((c + 1) & 1) * K2_BBUF, c + 1, tid, 4);
      wait_async_le4();
    } else {
      wait_async_0();
    }
    __syncthreads();

    const v16bf af = frag_ld(Apan + (wm * 16) * LDSA + c * KC, LDSA, lane);
    const __bf16* bb = reinterpret_cast<const __bf16*>(smem + (c & 1) * K2_BBUF)
                     + (wn * 64) * LDSB;
    v16bf bf[4];
#pragma unroll
    for (int t = 0; t < 4; ++t) bf[t] = frag_ld(bb + t * 16 * LDSB, LDSB, lane);
    if (c < 16) {
#pragma unroll
      for (int t = 0; t < 4; ++t)
        ax[t] = __builtin_amdgcn_wmma_f32_16x16x32_bf16(
            false, af, false, bf[t], (short)0, ax[t], false, false);
    } else {
#pragma unroll
      for (int t = 0; t < 4; ++t)
        ah[t] = __builtin_amdgcn_wmma_f32_16x16x32_bf16(
            false, af, false, bf[t], (short)0, ah[t], false, false);
    }
    __syncthreads();
  }

  // ---- epilogue -----------------------------------------------------------
  float* EX = reinterpret_cast<float*>(smem);
  float* EH = EX + 32 * 520;
  const int ldE = 520;
  const int r0 = wm * 16 + ((lane >> 4) << 3);
#pragma unroll
  for (int t = 0; t < 4; ++t) {
    const int col = wn * 64 + t * 16 + (lane & 15);
    const float b = bias[col];
#pragma unroll
    for (int v = 0; v < 8; ++v) {
      EX[(r0 + v) * ldE + col] = ax[t][v] + b;
      EH[(r0 + v) * ldE + col] = ah[t][v];
    }
  }
  __syncthreads();

#pragma unroll
  for (int rr = 0; rr < 2; ++rr) {
    const int row = wid * 2 + rr;
    const size_t mg = (size_t)(mBase + row);
    float pre[16], s1 = 0.f, s2 = 0.f;
#pragma unroll
    for (int j = 0; j < 16; ++j) {
      const int idx = lane + j * 32;
      const float r = (float)rz[mg * 1024 + idx];
      const float p = EX[row * ldE + idx] + r * EH[row * ldE + idx];
      pre[j] = p; s1 += p; s2 += p * p;
    }
#pragma unroll
    for (int o = 16; o; o >>= 1) {
      s1 += __shfl_xor(s1, o, 32);
      s2 += __shfl_xor(s2, o, 32);
    }
    const float mean = s1 * (1.f / 512.f);
    const float rstd = rsqrtf(s2 * (1.f / 512.f) - mean * mean + 1e-5f);
#pragma unroll
    for (int j = 0; j < 16; ++j) {
      const int idx = lane + j * 32;
      const float ht = tanhf((pre[j] - mean) * rstd * gam[idx] + bet[idx]);
      const float z  = (float)rz[mg * 1024 + DIM + idx];
      const float hp = hprev[mg * DIM + idx];
      out[mg * DIM + idx] = (1.f - z) * hp + z * ht;
    }
  }
}

// ===========================================================================
extern "C" void kernel_launch(void* const* d_in, const int* in_sizes, int n_in,
                              void* d_out, int out_size, void* d_ws, size_t ws_size,
                              hipStream_t stream) {
  (void)in_sizes; (void)n_in; (void)out_size; (void)ws_size;
  const float* x       = (const float*)d_in[0];
  const float* h       = (const float*)d_in[1];
  const float* w_rz_w  = (const float*)d_in[2];
  const float* w_rz_b  = (const float*)d_in[3];
  const float* u_rz_w  = (const float*)d_in[4];
  const float* w_h_w   = (const float*)d_in[5];
  const float* w_h_b   = (const float*)d_in[6];
  const float* u_h_w   = (const float*)d_in[7];
  const float* ln_rz_g = (const float*)d_in[8];
  const float* ln_rz_b = (const float*)d_in[9];
  const float* ln_h_g  = (const float*)d_in[10];
  const float* ln_h_b  = (const float*)d_in[11];
  float* out = (float*)d_out;

  char* ws = (char*)d_ws;
  __bf16* Wrz = reinterpret_cast<__bf16*>(ws);                             // 2 MB
  __bf16* Wh  = reinterpret_cast<__bf16*>(ws + (size_t)2 * 1024 * 1024);   // 1 MB
  __bf16* rz  = reinterpret_cast<__bf16*>(ws + (size_t)3 * 1024 * 1024);   // 128 MB

  prepack_weights<<<6144, 256, 0, stream>>>(w_rz_w, u_rz_w, w_h_w, u_h_w, Wrz, Wh);
  rz_gemm_ln<<<Bsz / MT, NTHR, K1_SMEM, stream>>>(x, h, Wrz, w_rz_b,
                                                  ln_rz_g, ln_rz_b, rz);
  h_gemm_out<<<Bsz / MT, NTHR, K2_SMEM, stream>>>(x, h, Wh, w_h_b,
                                                  ln_h_g, ln_h_b, rz, out);
}
